// PSA_53060025974836
// MI455X (gfx1250) — matrix operand order repaired and checked by reading, not compiled, not run on hardware
//
#include <hip/hip_runtime.h>
#include <hip/hip_bf16.h>

typedef _Float16 v16h __attribute__((ext_vector_type(16)));
typedef _Float16 v8h  __attribute__((ext_vector_type(8)));
typedef float    v8f  __attribute__((ext_vector_type(8)));

#define WMMA16(a,b,c) __builtin_amdgcn_wmma_f32_16x16x32_f16(false,(a),false,(b),(short)0,(c),false,false)

static constexpr int BATCH = 4;
static constexpr int CH    = 256;
static constexpr int NN    = 4096;   // 64*64
static constexpr int KD    = 32;     // CH/8
static constexpr float SCALE = 0.17677669529663687f;   // 1/sqrt(32)
static constexpr float LOG2E = 1.4426950408889634f;

// workspace layout (bytes, all 256-aligned)
static constexpr size_t OFF_W16 = 0;                       // [2][320][256] f16
static constexpr size_t OFF_FT  = 327680;                  // [2][4][4096][32] f16
static constexpr size_t OFF_GT  = OFF_FT  + 2097152;       // [2][4][4096][32] f16
static constexpr size_t OFF_L   = OFF_GT  + 2097152;       // [2][4][4096]     f32
static constexpr size_t OFF_H16 = OFF_L   + 131072;        // [2][4][256][4096] f16
static constexpr size_t OFF_XT  = OFF_H16 + 16777216;      // [2][4][4096][256] f16
// total = 38,207,488 bytes

// ---- fragment loaders (ISA 16-bit A/B VGPR layouts) ----
// A-matrix 16x32: lane holds row M=l%16; halves: K = 8*hl+{0..7} and 16+8*hl+{0..7}
static __device__ inline v16h ldA(const _Float16* row, int hl) {
    v8h a = *(const v8h*)(row + 8 * hl);
    v8h b = *(const v8h*)(row + 16 + 8 * hl);
    v16h r;
#pragma unroll
    for (int i = 0; i < 8; ++i) { r[i] = a[i]; r[i + 8] = b[i]; }
    return r;
}
// B-matrix 32x16: lane holds col N=l%16; halves: K = 8*hl+{0..7} and 16+8*hl+{0..7}
static __device__ inline v16h ldB(const _Float16* row, int hl) {
    v8h a = *(const v8h*)(row + 8 * hl);
    v8h b = *(const v8h*)(row + 16 + 8 * hl);
    v16h r;
#pragma unroll
    for (int i = 0; i < 8; ++i) { r[i] = a[i]; r[i + 8] = b[i]; }
    return r;
}

// ---------------- K0a: weights -> f16, concat [f;g;h_br] = [320][256] per branch ----
__global__ void k0a_weights(const float* __restrict__ f_w, const float* __restrict__ g_w,
                            const float* __restrict__ h1_w, const float* __restrict__ h2_w,
                            _Float16* __restrict__ W16) {
    int idx = blockIdx.x * 256 + threadIdx.x;
    if (idx >= 2 * 320 * 256) return;
    int c  = idx & 255;
    int o  = (idx >> 8) % 320;
    int br = idx / (320 * 256);
    float w = (o < 32)  ? f_w[o * 256 + c]
            : (o < 64)  ? g_w[(o - 32) * 256 + c]
            : (br ? h2_w : h1_w)[(o - 64) * 256 + c];
    W16[idx] = (_Float16)w;
}

// ---------------- K0b: x [b][c][n] f32 -> XT [b][n][c] f16 (LDS tiled transpose) ----
__global__ void k0b_transpose(const float* __restrict__ x1, const float* __restrict__ x2,
                              _Float16* __restrict__ XT) {
    __shared__ _Float16 t[32][33];
    int zb = blockIdx.z;                 // br*4 + b
    int br = zb >> 2, b = zb & 3;
    const float* xb = (br ? x2 : x1) + (size_t)b * CH * NN;
    int n0 = blockIdx.x * 32, c0 = blockIdx.y * 32;
#pragma unroll
    for (int i = 0; i < 4; ++i) {
        int c = c0 + threadIdx.y + i * 8;
        t[threadIdx.y + i * 8][threadIdx.x] = (_Float16)xb[(size_t)c * NN + n0 + threadIdx.x];
    }
    __syncthreads();
    _Float16* XTb = XT + (size_t)zb * NN * CH;
#pragma unroll
    for (int i = 0; i < 4; ++i) {
        int n = n0 + threadIdx.y + i * 8;
        XTb[(size_t)n * CH + c0 + threadIdx.x] = t[threadIdx.x][threadIdx.y + i * 8];
    }
}

// ---------------- K1: projection P = W @ X  (WMMA), scatter to FT/GT/H16 ------------
__global__ void k1_project(const _Float16* __restrict__ W16, const _Float16* __restrict__ XT,
                           _Float16* __restrict__ FT, _Float16* __restrict__ GT,
                           _Float16* __restrict__ H16,
                           const float* __restrict__ f_b, const float* __restrict__ g_b,
                           const float* __restrict__ h1_b, const float* __restrict__ h2_b) {
    int tid  = threadIdx.x;
    int lane = tid & 31, li = lane & 15, hl = lane >> 4;
    int wid  = (blockIdx.x * blockDim.x + tid) >> 5;       // 0..10239
    int og   = wid % 5;
    int rest = wid / 5;
    int nt = rest & 255, b = (rest >> 8) & 3, br = rest >> 10;

    const _Float16* XTb = XT + (size_t)(br * 4 + b) * NN * CH;
    const float* hb = br ? h2_b : h1_b;

    v8f acc[4];
#pragma unroll
    for (int i = 0; i < 4; ++i) {
        int ob = (og * 4 + i) * 16;
#pragma unroll
        for (int v = 0; v < 8; ++v) {
            int o = ob + v + 8 * hl;
            acc[i][v] = (o < 32) ? f_b[o] : (o < 64) ? g_b[o - 32] : hb[o - 64];
        }
    }
    int n = nt * 16 + li;
#pragma unroll
    for (int kc = 0; kc < 8; ++kc) {
        int c0 = kc * 32;
        v16h bx = ldB(XTb + (size_t)n * CH + c0, hl);      // X fragment, reused by 4 o-tiles
#pragma unroll
        for (int i = 0; i < 4; ++i) {
            int ob = (og * 4 + i) * 16;
            v16h aw = ldA(W16 + ((size_t)br * 320 + ob + li) * CH + c0, hl);
            acc[i] = WMMA16(aw, bx, acc[i]);
        }
    }
#pragma unroll
    for (int i = 0; i < 4; ++i) {
        int ob = (og * 4 + i) * 16;
        if (ob < 64) {                                      // F (ob<32) or G tiles
            _Float16* base = (ob >= 32 ? GT : FT);
            _Float16* dst = base + ((size_t)(br * 4 + b) * NN + n) * KD + (ob & 16) + 8 * hl;
            v8h pk;
#pragma unroll
            for (int v = 0; v < 8; ++v) pk[v] = (_Float16)acc[i][v];
            *(v8h*)dst = pk;
        } else {                                            // H: [c][n] layout
#pragma unroll
            for (int v = 0; v < 8; ++v) {
                int c = ob - 64 + v + 8 * hl;
                H16[((size_t)(br * 4 + b) * CH + c) * NN + n] = (_Float16)acc[i][v];
            }
        }
    }
}

// ---------------- K2: per-row (n) logsumexp over m of scale*fᵀg --------------------
__global__ void k2_rowstats(const _Float16* __restrict__ FT, const _Float16* __restrict__ GT,
                            float* __restrict__ Lc) {
    int tid  = threadIdx.x;
    int lane = tid & 31, li = lane & 15, hl = lane >> 4;
    int wid  = (blockIdx.x * blockDim.x + tid) >> 5;       // 0..2047
    int nt = wid & 255, b = (wid >> 8) & 3, br = wid >> 10;

    const _Float16* FTb = FT + (size_t)(br * 4 + b) * NN * KD;
    const _Float16* GTb = GT + (size_t)(br * 4 + b) * NN * KD;

    v16h bf = ldB(FTb + (size_t)(nt * 16 + li) * KD, hl);  // F fragment: loop-invariant
    float rmax = -1e30f, rsum = 0.0f;

    for (int mt = 0; mt < 256; ++mt) {
        v16h ag = ldA(GTb + (size_t)(mt * 16 + li) * KD, hl);
        v8f s = {};
        s = WMMA16(ag, bf, s);                             // D[m_loc][n_loc]; lane owns col n
        float tm = s[0];
#pragma unroll
        for (int v = 1; v < 8; ++v) tm = fmaxf(tm, s[v]);
        float nm = fmaxf(rmax, tm);
        float ts = 0.0f;
#pragma unroll
        for (int v = 0; v < 8; ++v) ts += __expf(SCALE * (s[v] - nm));
        rsum = rsum * __expf(SCALE * (rmax - nm)) + ts;
        rmax = nm;
    }
    float om = __shfl_xor(rmax, 16);
    float os = __shfl_xor(rsum, 16);
    float nm = fmaxf(rmax, om);
    float tot = rsum * __expf(SCALE * (rmax - nm)) + os * __expf(SCALE * (om - nm));
    if (lane < 16)
        Lc[(size_t)(br * 4 + b) * NN + nt * 16 + lane] = (SCALE * nm + __logf(tot)) * LOG2E;
}

// ---------------- K3: sa_map = H * exp2(scale*log2e*s - L[n]);  out = gamma*sa + x --
// Wave pairs cooperate: each wave of a pair computes one 16-n score subtile (1 WMMA),
// E tile shared via double-buffered LDS; both waves then run 8 main WMMAs each.
__global__ void k3_attn_gemm(const _Float16* __restrict__ FT, const _Float16* __restrict__ GT,
                             const _Float16* __restrict__ H16, const float* __restrict__ Lc,
                             const float* __restrict__ x1, const float* __restrict__ x2,
                             const float* __restrict__ gamma1, const float* __restrict__ gamma2,
                             float* __restrict__ out) {
    __shared__ __align__(16) _Float16 esm[4][2][16 * 40];  // per-pair double-buffered E tile
    int tid  = threadIdx.x;
    int lane = tid & 31, li = lane & 15, hl = lane >> 4;
    int wv   = tid >> 5;                                   // 0..7
    int pr   = wv >> 1;                                    // pair in block: 0..3
    int cg   = wv & 1;                                     // channel half / E subtile owner
    int pid  = blockIdx.x * 4 + pr;                        // 0..2047
    int mt = pid & 255, b = (pid >> 8) & 3, br = (pid >> 10) & 1;

    size_t bb = (size_t)(br * 4 + b);
    const _Float16* FTb = FT  + bb * NN * KD;
    const _Float16* GTb = GT  + bb * NN * KD;
    const _Float16* Hb  = H16 + bb * CH * NN;
    const float*    Lb  = Lc  + bb * NN;

    v16h bg = ldB(GTb + (size_t)(mt * 16 + li) * KD, hl);  // G fragment: loop-invariant
    v8f acc[8];
#pragma unroll
    for (int i = 0; i < 8; ++i) acc[i] = (v8f){};

    const float k1 = SCALE * LOG2E;
    for (int nc = 0; nc < 128; ++nc) {
        int nb = nc * 32;
        _Float16* E = esm[pr][nc & 1];
        // this wave computes score subtile s = cg  (n0 .. n0+15)
        {
            int n0 = nb + cg * 16;
            v16h af = ldA(FTb + (size_t)(n0 + li) * KD, hl);
            v8f sc = {};
            sc = WMMA16(af, bg, sc);                       // D[n_loc][m]; lane owns col m
            float4 l0 = *(const float4*)(Lb + n0 + 8 * hl);
            float4 l1 = *(const float4*)(Lb + n0 + 8 * hl + 4);
            v8h e;
            e[0] = (_Float16)exp2f(sc[0] * k1 - l0.x);
            e[1] = (_Float16)exp2f(sc[1] * k1 - l0.y);
            e[2] = (_Float16)exp2f(sc[2] * k1 - l0.z);
            e[3] = (_Float16)exp2f(sc[3] * k1 - l0.w);
            e[4] = (_Float16)exp2f(sc[4] * k1 - l1.x);
            e[5] = (_Float16)exp2f(sc[5] * k1 - l1.y);
            e[6] = (_Float16)exp2f(sc[6] * k1 - l1.z);
            e[7] = (_Float16)exp2f(sc[7] * k1 - l1.w);
            *(v8h*)(E + li * 40 + 8 * hl + 16 * cg) = e;   // [m][n] staging
        }
        __syncthreads();                                   // waits own DScnt + block barrier
        v16h be = ldB(E + li * 40, hl);                    // full 32-n E as B fragment
#pragma unroll
        for (int i = 0; i < 8; ++i) {
            int c = cg * 128 + i * 16 + li;
            v16h ah = ldA(Hb + (size_t)c * NN + nb, hl);
            acc[i] = WMMA16(ah, be, acc[i]);
        }
    }
    // epilogue: out = gamma * sa_map + x
    float g = (br ? gamma2 : gamma1)[0];
    const float* xb = (br ? x2 : x1) + (size_t)b * CH * NN;
    float* ob = out + bb * CH * NN;
    int m = mt * 16 + li;
#pragma unroll
    for (int i = 0; i < 8; ++i) {
#pragma unroll
        for (int v = 0; v < 8; ++v) {
            int c = cg * 128 + i * 16 + v + 8 * hl;
            size_t off = (size_t)c * NN + m;
            ob[off] = g * acc[i][v] + xb[off];
        }
    }
}

extern "C" void kernel_launch(void* const* d_in, const int* in_sizes, int n_in,
                              void* d_out, int out_size, void* d_ws, size_t ws_size,
                              hipStream_t stream) {
    (void)in_sizes; (void)n_in; (void)out_size; (void)ws_size;
    const float* x1    = (const float*)d_in[0];
    const float* x2    = (const float*)d_in[1];
    const float* f_w   = (const float*)d_in[2];
    const float* f_b   = (const float*)d_in[3];
    const float* g_w   = (const float*)d_in[4];
    const float* g_b   = (const float*)d_in[5];
    const float* h1_w  = (const float*)d_in[6];
    const float* h1_b  = (const float*)d_in[7];
    const float* h2_w  = (const float*)d_in[8];
    const float* h2_b  = (const float*)d_in[9];
    const float* gam1  = (const float*)d_in[10];
    const float* gam2  = (const float*)d_in[11];
    float* out = (float*)d_out;

    char* ws = (char*)d_ws;
    _Float16* W16 = (_Float16*)(ws + OFF_W16);
    _Float16* FT  = (_Float16*)(ws + OFF_FT);
    _Float16* GT  = (_Float16*)(ws + OFF_GT);
    float*    Lc  = (float*)   (ws + OFF_L);
    _Float16* H16 = (_Float16*)(ws + OFF_H16);
    _Float16* XT  = (_Float16*)(ws + OFF_XT);

    k0a_weights<<<640, 256, 0, stream>>>(f_w, g_w, h1_w, h2_w, W16);
    k0b_transpose<<<dim3(128, 8, 8), dim3(32, 8), 0, stream>>>(x1, x2, XT);
    k1_project<<<1280, 256, 0, stream>>>(W16, XT, FT, GT, H16, f_b, g_b, h1_b, h2_b);
    k2_rowstats<<<256, 256, 0, stream>>>(FT, GT, Lc);
    k3_attn_gemm<<<512, 256, 0, stream>>>(FT, GT, H16, Lc, x1, x2, gam1, gam2, out);
}
